// Decoder_41850161332763
// MI455X (gfx1250) — matrix operand order repaired and checked by reading, not compiled, not run on hardware
//
#include <hip/hip_runtime.h>
#include <hip/hip_bf16.h>

// ---------------------------------------------------------------------------
// Pointer-network decoder, fused for MI455X (gfx1250, wave32, WMMA).
//   B=512, S=64, E=128, H=128
// Outputs (concat): cell_log_prob[512], cell_reward[512], cell_action[512*64]
// Round 3: fix async-to-LDS builtin arg types (v4i* + addrspace typedefs).
// Loop-invariant W1/Wv staged as f16 in LDS (CDNA5: 320KB/WGP) so per-step
// WMMA B-fragments are ds_load instead of global f32 + convert.
// ---------------------------------------------------------------------------

#define Bsz 512
#define Sl  64
#define Ed  128
#define Hd  128
#define ROWS 16           // batch rows per workgroup
#define C_CLIP 10.0f
#define MASK_NEG 1e8f

typedef __attribute__((ext_vector_type(16))) _Float16 v16h;
typedef __attribute__((ext_vector_type(8)))  float    v8f;
typedef __attribute__((ext_vector_type(4)))  int      v4i;
typedef __attribute__((address_space(1))) v4i* as1_v4i;
typedef __attribute__((address_space(3))) v4i* as3_v4i;

#if __has_builtin(__builtin_amdgcn_global_load_async_to_lds_b128) && \
    __has_builtin(__builtin_amdgcn_s_wait_asynccnt)
#define HAVE_ASYNC_LDS 1
#else
#define HAVE_ASYNC_LDS 0
#endif

// ---- WMMA fragment loaders (ISA 7.12.2 layouts, wave32) --------------------
// A: 16x32 f16.  lane L: M = L&15, koff = (L<16)?0:8
//   V0..V3 : K = koff+{0..7};  V4..V7 : K = 16+koff+{0..7}
__device__ __forceinline__ v16h load_a_f16(const _Float16* base, int ld,
                                           int kbase, int lane) {
  const int m    = lane & 15;
  const int koff = (lane < 16) ? 0 : 8;
  const _Float16* row = base + m * ld + kbase;
  v16h a;
#pragma unroll
  for (int j = 0; j < 4; ++j) {
    a[2 * j]     = row[koff + 2 * j];
    a[2 * j + 1] = row[koff + 2 * j + 1];
    a[8 + 2 * j]     = row[16 + koff + 2 * j];
    a[8 + 2 * j + 1] = row[16 + koff + 2 * j + 1];
  }
  return a;
}

// A fragment directly from global f32 (converted to f16)
__device__ __forceinline__ v16h load_a_glb_f32(const float* base, int ld,
                                               int kbase, int lane) {
  const int m    = lane & 15;
  const int koff = (lane < 16) ? 0 : 8;
  const float* row = base + (size_t)m * ld + kbase;
  v16h a;
#pragma unroll
  for (int j = 0; j < 4; ++j) {
    a[2 * j]     = (_Float16)row[koff + 2 * j];
    a[2 * j + 1] = (_Float16)row[koff + 2 * j + 1];
    a[8 + 2 * j]     = (_Float16)row[16 + koff + 2 * j];
    a[8 + 2 * j + 1] = (_Float16)row[16 + koff + 2 * j + 1];
  }
  return a;
}

// B: 32x16 f16 (K x N), f16 LDS weight W[k*ldn + n]  -> ds_load path
//   lane L: N = n0 + (L&15), kb = (L<16)?0:8 ; V0..V3: K = kb+{0..7}, V4..V7: K = 16+kb+{0..7}
__device__ __forceinline__ v16h load_b_lds(const _Float16* W, int ldn,
                                           int kbase, int n0, int lane) {
  const int n  = n0 + (lane & 15);
  const int kb = (lane < 16) ? 0 : 8;
  v16h b;
#pragma unroll
  for (int j = 0; j < 8; ++j) {
    b[2 * j]     = W[(kbase + kb + 2 * j + (j & 4) * 2) * ldn + n];
    b[2 * j + 1] = W[(kbase + kb + 2 * j + (j & 4) * 2 + 1) * ldn + n];
  }
  return b;
}

// B fragment from row-major f32 global weight (one-shot GEMM kernel only)
__device__ __forceinline__ v16h load_b_glb_f32(const float* W, int ldn,
                                               int kbase, int n0, int lane) {
  const int n  = n0 + (lane & 15);
  const int kb = (lane < 16) ? 0 : 8;
  v16h b;
#pragma unroll
  for (int j = 0; j < 8; ++j) {
    b[2 * j]     = (_Float16)W[(size_t)(kbase + kb + 2 * j + (j & 4) * 2) * ldn + n];
    b[2 * j + 1] = (_Float16)W[(size_t)(kbase + kb + 2 * j + (j & 4) * 2 + 1) * ldn + n];
  }
  return b;
}

// deterministic per-(step,row) uniform in (0,1)
__device__ __forceinline__ float u01_hash(unsigned a, unsigned b) {
  unsigned x = a * 0x9E3779B9u ^ (b + 0x85EBCA6Bu + (a << 6) + (a >> 2));
  x ^= x >> 16; x *= 0x7FEB352Du;
  x ^= x >> 15; x *= 0x846CA68Bu;
  x ^= x >> 16;
  return (float)(x >> 8) * (1.0f / 16777216.0f) + 1e-7f;
}

// ---------------------------------------------------------------------------
// Kernel A: h_bar = mean_s(cc) @ Wc + bc          (one block per batch row)
// ---------------------------------------------------------------------------
__global__ void prep_hbar_kernel(const float* __restrict__ cc,
                                 const float* __restrict__ Wc,
                                 const float* __restrict__ bc,
                                 float* __restrict__ hbar) {
  __shared__ float hm[Ed];
  const int b = blockIdx.x, e = threadIdx.x;
  float acc = 0.f;
  for (int s = 0; s < Sl; ++s) acc += cc[((size_t)b * Sl + s) * Ed + e];
  hm[e] = acc * (1.0f / (float)Sl);
  __syncthreads();
  float o = bc[e];
  for (int k = 0; k < Ed; ++k) o += hm[k] * Wc[k * Ed + e];
  hbar[(size_t)b * Ed + e] = o;
}

// ---------------------------------------------------------------------------
// Kernel A2: wvinit = init_w @ Wv + bv            (single block)
// ---------------------------------------------------------------------------
__global__ void prep_wvinit_kernel(const float* __restrict__ init_w,
                                   const float* __restrict__ Wv,
                                   const float* __restrict__ bv,
                                   float* __restrict__ wvinit) {
  const int e = threadIdx.x;
  float o = bv[e];
  for (int k = 0; k < 2 * Ed; ++k) o += init_w[k] * Wv[k * Ed + e];
  wvinit[e] = o;
}

// ---------------------------------------------------------------------------
// Kernel B: tgt = cc @ W2 + b2, stored f16.  M=32768 tiles of 16, 8 waves
// cover the 8 N-tiles.  K=128 -> 4 chained v_wmma_f32_16x16x32_f16.
// ---------------------------------------------------------------------------
__global__ __launch_bounds__(256) void tgt_gemm_kernel(
    const float* __restrict__ cc, const float* __restrict__ W2,
    const float* __restrict__ b2, _Float16* __restrict__ tgt) {
  const int wave = threadIdx.x >> 5, lane = threadIdx.x & 31;
  const int m0 = blockIdx.x * 16;
  const int n0 = wave * 16;
  v8f c = {};
#pragma unroll
  for (int kc = 0; kc < 4; ++kc) {
    v16h a = load_a_glb_f32(cc + (size_t)m0 * Ed, Ed, kc * 32, lane);
    v16h b = load_b_glb_f32(W2, Hd, kc * 32, n0, lane);
    c = __builtin_amdgcn_wmma_f32_16x16x32_f16(false, a, false, b,
                                               (short)0, c, false, false);
  }
  const int n = n0 + (lane & 15);
  const int moff = (lane < 16) ? 0 : 8;
  const float bias = b2[n];
#pragma unroll
  for (int r = 0; r < 8; ++r)
    tgt[(size_t)(m0 + moff + r) * Hd + n] = (_Float16)(c[r] + bias);
}

// ---------------------------------------------------------------------------
// Kernel C: persistent fused decode.  32 blocks x 256 threads; each block
// owns 16 independent batch rows for all 64 steps.  All loop-carried state
// and both loop-invariant weight matrices live in LDS (~135KB of 320KB).
// ---------------------------------------------------------------------------
__global__ __launch_bounds__(256) void decode_kernel(
    const float* __restrict__ cc, const float* __restrict__ od,
    const float* __restrict__ highmask,
    const float* __restrict__ Wv, const float* __restrict__ bv,
    const float* __restrict__ W1, const float* __restrict__ b1,
    const float* __restrict__ wv, const float* __restrict__ bvp,
    const float* __restrict__ hbar_g, const float* __restrict__ wvinit,
    const _Float16* __restrict__ tgt, float* __restrict__ out) {
  __shared__ _Float16 W1h[Ed * Hd];          // 32KB  f16 W1 (B operand)
  __shared__ _Float16 Wvh[2 * Ed * Ed];      // 64KB  f16 Wv (B operand)
  __shared__ _Float16 queryh[ROWS][Ed];      // current query (f16, WMMA A)
  __shared__ _Float16 cath[ROWS][2 * Ed];    // concat(init_h, h) (f16, WMMA A)
  __shared__ float qbuf[ROWS][Hd];           // q = query@W1 + b1
  __shared__ float hbar_s[ROWS][Ed];
  __shared__ float ubuf[ROWS][Sl];           // u, then logits in-place
  __shared__ float maskf[ROWS][Sl];
  __shared__ float wv_s[Hd], b1_s[Hd], bv_s[Ed];
  __shared__ int   idx_s[ROWS];
  __shared__ float logp_s[ROWS], rew_s[ROWS], lastx[ROWS], lasty[ROWS];

  const int tid  = threadIdx.x;
  const int wave = tid >> 5, lane = tid & 31;
  const int b0 = blockIdx.x * ROWS;

  // ---- one-time staging ----------------------------------------------------
#if HAVE_ASYNC_LDS
  // small vectors via CDNA5 async global->LDS DMA (ASYNCcnt path)
  if (tid < Hd / 4) {
    __builtin_amdgcn_global_load_async_to_lds_b128(
        (as1_v4i)(v4i*)(wv + tid * 4), (as3_v4i)(v4i*)(wv_s + tid * 4), 0, 0);
    __builtin_amdgcn_global_load_async_to_lds_b128(
        (as1_v4i)(v4i*)(b1 + tid * 4), (as3_v4i)(v4i*)(b1_s + tid * 4), 0, 0);
    __builtin_amdgcn_global_load_async_to_lds_b128(
        (as1_v4i)(v4i*)(bv + tid * 4), (as3_v4i)(v4i*)(bv_s + tid * 4), 0, 0);
  }
  __builtin_amdgcn_s_wait_asynccnt(0);
#else
  if (tid < Hd) { wv_s[tid] = wv[tid]; b1_s[tid] = b1[tid]; bv_s[tid] = bv[tid]; }
#endif
  for (int i = tid; i < Ed * Hd; i += 256) W1h[i] = (_Float16)W1[i];
  for (int i = tid; i < 2 * Ed * Ed; i += 256) Wvh[i] = (_Float16)Wv[i];
  for (int i = tid; i < ROWS * Ed; i += 256) {
    const int m = i >> 7, e = i & 127;
    const float hb = hbar_g[(size_t)(b0 + m) * Ed + e];
    hbar_s[m][e] = hb;
    queryh[m][e] = (_Float16)(hb + wvinit[e]);
    cath[m][e] = (_Float16)0.f;
    cath[m][Ed + e] = (_Float16)0.f;
  }
  for (int i = tid; i < ROWS * Sl; i += 256) {
    const int m = i >> 6, s = i & 63;
    maskf[m][s] = highmask[(size_t)(b0 + m) * Sl + s];
  }
  if (tid < ROWS) {
    idx_s[tid] = 0; logp_s[tid] = 0.f; rew_s[tid] = 0.f;
    lastx[tid] = 0.f; lasty[tid] = 0.f;
  }
  __syncthreads();

  const float bvp0 = bvp[0];

  for (int step = 0; step < Sl; ++step) {
    // -- phase 1: q = query @ W1 + b1  (wave w -> N-tile w; 4 WMMAs) --------
    {
      const int n0 = wave * 16;
      v8f c = {};
#pragma unroll
      for (int kc = 0; kc < 4; ++kc) {
        v16h a = load_a_f16(&queryh[0][0], Ed, kc * 32, lane);
        v16h b = load_b_lds(W1h, Hd, kc * 32, n0, lane);
        c = __builtin_amdgcn_wmma_f32_16x16x32_f16(false, a, false, b,
                                                   (short)0, c, false, false);
      }
      const int n = n0 + (lane & 15);
      const int moff = (lane < 16) ? 0 : 8;
      const float bias = b1_s[n];
#pragma unroll
      for (int r = 0; r < 8; ++r) qbuf[moff + r][n] = c[r] + bias;
      // overlap: prefetch this step's tgt tile (L2-resident stream)
      const _Float16* p =
          tgt + ((size_t)(b0 + (tid >> 4)) * Sl + (size_t)(tid & 15) * 4) * Hd;
      __builtin_prefetch((const void*)p, 0, 1);
    }
    __syncthreads();

    // -- phase 2: u[b,s] = tanh(q + tgt) . wv + bvp  (2 rows per wave) ------
    {
      const int row = wave * 2 + (lane >> 4);
      const int lg = lane & 15;
      const _Float16* trow = tgt + (size_t)(b0 + row) * Sl * Hd;
      for (int s = 0; s < Sl; ++s) {
        float acc = 0.f;
#pragma unroll
        for (int jj = 0; jj < 8; ++jj) {
          const int h = lg * 8 + jj;
          acc += tanhf(qbuf[row][h] + (float)trow[s * Hd + h]) * wv_s[h];
        }
        acc += __shfl_xor(acc, 8, 16);
        acc += __shfl_xor(acc, 4, 16);
        acc += __shfl_xor(acc, 2, 16);
        acc += __shfl_xor(acc, 1, 16);
        if (lg == 0) ubuf[row][s] = acc + bvp0;
      }
    }
    __syncthreads();

    // -- phase 3: clip + mask, categorical sample, logp, reward -------------
    if (tid < ROWS) {
      const int row = tid;
      float mx = -3.4e38f;
      for (int s = 0; s < Sl; ++s) {
        const float l = C_CLIP * tanhf(ubuf[row][s]) - MASK_NEG * maskf[row][s];
        ubuf[row][s] = l;                       // logits in-place
        mx = fmaxf(mx, l);
      }
      float se = 0.f;
      for (int s = 0; s < Sl; ++s) se += __expf(ubuf[row][s] - mx);
      const float u01 = u01_hash(0xB5297A4Du + (unsigned)step,
                                 (unsigned)(b0 + row));
      const float target = u01 * se;
      float csum = 0.f;
      int idx = Sl - 1;
      for (int s = 0; s < Sl; ++s) {
        csum += __expf(ubuf[row][s] - mx);
        if (csum >= target) { idx = s; break; }
      }
      logp_s[row] += ubuf[row][idx] - mx - __logf(se);
      maskf[row][idx] = 1.f;
      idx_s[row] = idx;
      const float nx = od[((size_t)(b0 + row) * Sl + idx) * 2 + 0];
      const float ny = od[((size_t)(b0 + row) * Sl + idx) * 2 + 1];
      if (step > 0) {
        const float dx = nx - lastx[row], dy = ny - lasty[row];
        rew_s[row] += sqrtf(dx * dx + dy * dy);
      }
      lastx[row] = nx; lasty[row] = ny;
      out[2 * Bsz + (size_t)(b0 + row) * Sl + step] = (float)idx;
    }
    __syncthreads();

    // -- phase 4: gather h = cc[b, idx]; freeze init_h at step 0 ------------
    {
      const int row = tid >> 4, lg = tid & 15;
      const int idx = idx_s[row];
      const float* hrow = cc + ((size_t)(b0 + row) * Sl + idx) * Ed;
#pragma unroll
      for (int jj = 0; jj < 8; ++jj) {
        const int e = lg * 8 + jj;
        const _Float16 v = (_Float16)hrow[e];
        cath[row][Ed + e] = v;
        if (step == 0) cath[row][e] = v;
      }
    }
    __syncthreads();

    // -- phase 5: new_query = hbar + cat(init_h,h) @ Wv + bv  (8 WMMAs) -----
    {
      const int n0 = wave * 16;
      v8f c = {};
#pragma unroll
      for (int kc = 0; kc < 8; ++kc) {
        v16h a = load_a_f16(&cath[0][0], 2 * Ed, kc * 32, lane);
        v16h b = load_b_lds(Wvh, Ed, kc * 32, n0, lane);
        c = __builtin_amdgcn_wmma_f32_16x16x32_f16(false, a, false, b,
                                                   (short)0, c, false, false);
      }
      const int n = n0 + (lane & 15);
      const int moff = (lane < 16) ? 0 : 8;
#pragma unroll
      for (int r = 0; r < 8; ++r) {
        const int m = moff + r;
        queryh[m][n] = (_Float16)(hbar_s[m][n] + c[r] + bv_s[n]);
      }
    }
    __syncthreads();
  }

  if (tid < ROWS) {
    out[b0 + tid] = logp_s[tid];
    out[Bsz + b0 + tid] = rew_s[tid];
  }
}

// ---------------------------------------------------------------------------
extern "C" void kernel_launch(void* const* d_in, const int* in_sizes, int n_in,
                              void* d_out, int out_size, void* d_ws,
                              size_t ws_size, hipStream_t stream) {
  const float* cc       = (const float*)d_in[0];   // [B,S,E]
  const float* od       = (const float*)d_in[1];   // [B,S,2]
  const float* highmask = (const float*)d_in[2];   // [B,S]
  const float* init_w   = (const float*)d_in[3];   // [2E]
  const float* Wc       = (const float*)d_in[4];   // [E,E]
  const float* bc       = (const float*)d_in[5];   // [E]
  const float* Wv       = (const float*)d_in[6];   // [2E,E]
  const float* bv       = (const float*)d_in[7];   // [E]
  const float* W1       = (const float*)d_in[8];   // [E,H]
  const float* b1       = (const float*)d_in[9];   // [H]
  const float* W2       = (const float*)d_in[10];  // [E,H]
  const float* b2       = (const float*)d_in[11];  // [H]
  const float* wv       = (const float*)d_in[12];  // [H]
  const float* bvp      = (const float*)d_in[13];  // scalar
  float* out = (float*)d_out;

  // workspace layout
  char* ws = (char*)d_ws;
  float*     hbar_ws   = (float*)(ws);                       // 512*128 f32
  float*     wvinit_ws = (float*)(ws + 256 * 1024);          // 128 f32
  _Float16*  tgt_ws    = (_Float16*)(ws + 512 * 1024);       // 32768*128 f16

  prep_hbar_kernel<<<Bsz, Ed, 0, stream>>>(cc, Wc, bc, hbar_ws);
  prep_wvinit_kernel<<<1, Ed, 0, stream>>>(init_w, Wv, bv, wvinit_ws);
  tgt_gemm_kernel<<<(Bsz * Sl) / 16, 256, 0, stream>>>(cc, W2, b2, tgt_ws);
  decode_kernel<<<Bsz / ROWS, 256, 0, stream>>>(
      cc, od, highmask, Wv, bv, W1, b1, wv, bvp, hbar_ws, wvinit_ws, tgt_ws,
      out);
}
